// WinSelfAtt_76029511073925
// MI455X (gfx1250) — matrix operand order repaired and checked by reading, not compiled, not run on hardware
//
#include <hip/hip_runtime.h>
#include <hip/hip_bf16.h>

typedef __attribute__((ext_vector_type(16))) _Float16 v16h;
typedef __attribute__((ext_vector_type(8)))  _Float16 v8h;
typedef __attribute__((ext_vector_type(8)))  float    v8f;

#define NSEQ 430
#define SEQL 25
#define EMB  20
#define SVS  10
#define NHEAD 5
#define FFN  10
#define NLAY 4
#define BB   32
#define TA   256   // threads for encoder kernels (8 waves)
#define TB   64    // threads for linear kernel (2 waves)

// ---------------- WMMA fragment helpers ----------------
// A fragment (16x32): lane m = lane&15, kb = (lane>>4)*8 -> two 16B loads at p[0], p[16].
__device__ __forceinline__ v16h frag_a_lds(const _Float16* __restrict__ base, int ldh, int lane) {
  const _Float16* p = base + (lane & 15) * ldh + ((lane >> 4) << 3);
  v8h lo = *(const v8h*)p;
  v8h hi = *(const v8h*)(p + 16);
  return __builtin_shufflevector(lo, hi, 0, 1, 2, 3, 4, 5, 6, 7, 8, 9, 10, 11, 12, 13, 14, 15);
}

// B fragment (32x16) from row-major [n][k]: lane n = lane&15, kb = (lane>>4)*8 -> two 16B loads at p[0], p[16].
__device__ __forceinline__ v16h frag_b_lds(const _Float16* __restrict__ base, int ldh, int lane) {
  const _Float16* p = base + (lane & 15) * ldh + ((lane >> 4) << 3);
  v8h lo = *(const v8h*)p;
  v8h hi = *(const v8h*)(p + 16);
  return __builtin_shufflevector(lo, hi, 0, 1, 2, 3, 4, 5, 6, 7, 8, 9, 10, 11, 12, 13, 14, 15);
}

// D fragment store: element j -> M = j + (lane>>4)*8 ; N = lane&15
template <typename FD>
__device__ __forceinline__ void frag_store(const v8f& d, FD f, int lane) {
  const int n  = lane & 15;
  const int mb = (lane >> 4) << 3;
#pragma unroll
  for (int j = 0; j < 8; ++j) f(mb + j, n, d[j]);
}

__device__ __forceinline__ v8f wmma16(v16h a, v16h b, v8f c) {
  return __builtin_amdgcn_wmma_f32_16x16x32_f16(false, a, false, b, (short)0, c, false, false);
}

// stage one weight row (KV valid floats) into a zero-padded 32-half LDS row: 4x b128 stores
template <int KV>
__device__ __forceinline__ void stage_row(const float* __restrict__ src,
                                          _Float16* __restrict__ dst, bool rowvalid) {
  v8h r[4] = {{}, {}, {}, {}};
  if (rowvalid) {
#pragma unroll
    for (int k = 0; k < KV; ++k) r[k >> 3][k & 7] = (_Float16)src[k];
  }
  v8h* d = (v8h*)dst;
#pragma unroll
  for (int c = 0; c < 4; ++c) d[c] = r[c];
}

// in-place softmax over a 32-wide f16 row (one row per lane), vectorized LDS access
__device__ __forceinline__ void softmax_row32(_Float16* row) {
  v8h* rp = (v8h*)row;
  float r[BB];
#pragma unroll
  for (int c = 0; c < 4; ++c) {
    v8h ch = rp[c];
#pragma unroll
    for (int j = 0; j < 8; ++j) r[c * 8 + j] = (float)ch[j];
  }
  float mx = r[0];
#pragma unroll
  for (int t2 = 1; t2 < BB; ++t2) mx = fmaxf(mx, r[t2]);
  float sum = 0.0f;
#pragma unroll
  for (int t2 = 0; t2 < BB; ++t2) { r[t2] = __expf(r[t2] - mx); sum += r[t2]; }
  const float inv = 1.0f / sum;
#pragma unroll
  for (int c = 0; c < 4; ++c) {
    v8h ch;
#pragma unroll
    for (int j = 0; j < 8; ++j) ch[j] = (_Float16)(r[c * 8 + j] * inv);
    rp[c] = ch;
  }
}

// ---------------- Kernel A: per-(window, seq-pos) 4-layer encoder slab [S=32 x E=20] ----------------

__global__ __launch_bounds__(TA) void win_encoder_kernel(
    const float* __restrict__ X,
    const float* __restrict__ Wqkv, const float* __restrict__ bqkv,
    const float* __restrict__ Wo,   const float* __restrict__ bo,
    const float* __restrict__ W1,   const float* __restrict__ b1,
    const float* __restrict__ W2,   const float* __restrict__ b2,
    const float* __restrict__ g1,   const float* __restrict__ be1,
    const float* __restrict__ g2,   const float* __restrict__ be2,
    float* __restrict__ Y)
{
  const int i    = blockIdx.x;   // window index
  const int n    = blockIdx.y;   // seq position
  const int tid  = threadIdx.x;
  const int wave = tid >> 5;
  const int lane = tid & 31;
  const int srow = tid & 31;
  const int ecol = tid >> 5;

  __shared__ __align__(16) _Float16 xh[BB][32];            // activations f16 (cols < 20)
  __shared__ float                  xbf[BB][EMB];          // activations f32
  __shared__ __align__(16) _Float16 wh[64][32];            // staged weights [out][k], padded
  __shared__ float                  bst[64];               // staged bias, zero-padded
  __shared__ __align__(16) _Float16 qkv64[BB][64];         // raw qkv GEMM result
  __shared__ __align__(16) _Float16 qpad[NHEAD][BB][32];   // q per head (cols < 4)
  __shared__ __align__(16) _Float16 kpad[NHEAD][BB][32];   // k per head (cols < 4)
  __shared__ __align__(16) _Float16 vT[NHEAD][16][32];     // v transposed (rows < 4)
  __shared__ __align__(16) _Float16 scb[NHEAD][BB][32];    // scores then probs
  __shared__ __align__(16) _Float16 oh[BB][32];            // attention out f16 (cols < 20)
  __shared__ __align__(16) _Float16 hh[BB][32];            // FFN hidden f16 (cols < 16 written)
  __shared__ float                  pbuf[BB][32];          // f32 projection scratch, padded

  // zero-init padded f16 operand buffers (padding persists zero across layers)
  for (int t = tid; t < BB * 32; t += TA) { ((_Float16*)xh)[t] = (_Float16)0.0f;
                                            ((_Float16*)oh)[t] = (_Float16)0.0f;
                                            ((_Float16*)hh)[t] = (_Float16)0.0f; }
  for (int t = tid; t < NHEAD * BB * 32; t += TA) { ((_Float16*)qpad)[t] = (_Float16)0.0f;
                                                    ((_Float16*)kpad)[t] = (_Float16)0.0f; }
  for (int t = tid; t < NHEAD * 16 * 32; t += TA) ((_Float16*)vT)[t] = (_Float16)0.0f;

  // load slab + positional encoding over s(=b); div-free indexing
  for (int e = ecol; e < EMB; e += 8) {
    float v = X[(((size_t)srow * NSEQ + i) * SEQL + n) * EMB + e];
    const int p2 = e & ~1;
    const float div = __expf((float)p2 * (-__logf(10000.0f) / (float)EMB));
    const float ang = (float)srow * div;
    v += (e & 1) ? __cosf(ang) : __sinf(ang);
    xbf[srow][e] = v;
    xh[srow][e] = (_Float16)v;
  }
  __syncthreads();

#pragma unroll 1
  for (int l = 0; l < NLAY; ++l) {
    const size_t wb = (size_t)i * NLAY + l;
    if (l + 1 < NLAY) {  // global_prefetch_b8 for next layer's big weights
      __builtin_prefetch(Wqkv + (wb + 1) * 60 * EMB, 0, 1);
      __builtin_prefetch(Wo + (wb + 1) * EMB * EMB, 0, 1);
    }

    // ---- qkv = x @ Wqkv^T + bqkv : M=32, N=60(pad64), K=20(pad32); 8 tiles -> 8 waves ----
    {
      const float* wq = Wqkv + wb * 60 * EMB;
      const float* bq = bqkv + wb * 60;
      if (tid < 64) {
        stage_row<EMB>(wq + tid * EMB, &wh[tid][0], tid < 60);
        bst[tid] = (tid < 60) ? bq[tid] : 0.0f;
      }
      __syncthreads();
      {
        const int mt = wave & 1, nt = wave >> 1;
        v16h a = frag_a_lds(&xh[mt * 16][0], 32, lane);
        v16h b = frag_b_lds(&wh[nt * 16][0], 32, lane);
        v8f c = {};
        c = wmma16(a, b, c);
        frag_store(c, [&](int m, int nn, float v) {  // unconditional: padding cols compute to 0
          const int o = nt * 16 + nn;
          qkv64[mt * 16 + m][o] = (_Float16)(v + bst[o]); }, lane);
      }
      __syncthreads();
      // repack: q/k head slices are contiguous 8B groups; v needs a scalar transpose
      if (tid < NHEAD * BB) {
        const int h = tid >> 5, row = tid & 31;
        *(unsigned long long*)&qpad[h][row][0] =
            *(const unsigned long long*)&qkv64[row][h * 4];
        *(unsigned long long*)&kpad[h][row][0] =
            *(const unsigned long long*)&qkv64[row][EMB + h * 4];
      }
      {
        auto vcopy = [&](int t) {
          const int h = t >> 7, rem = t & 127, row = rem >> 2, d = rem & 3;
          vT[h][d][row] = qkv64[row][2 * EMB + h * 4 + d];
        };
        vcopy(tid);
        vcopy(tid + 256);
        if (tid < 128) vcopy(tid + 512);
      }
      __syncthreads();
    }

    // ---- attention per head (D=4): one head per wave ----
    if (wave < NHEAD) {
      const int h = wave;
#pragma unroll
      for (int mt = 0; mt < 2; ++mt)
#pragma unroll
        for (int nt = 0; nt < 2; ++nt) {
          v16h a = frag_a_lds(&qpad[h][mt * 16][0], 32, lane);
          v16h b = frag_b_lds(&kpad[h][nt * 16][0], 32, lane);
          v8f c = {};
          c = wmma16(a, b, c);
          frag_store(c, [&](int m, int nn, float v) {
            scb[h][mt * 16 + m][nt * 16 + nn] = (_Float16)(v * 0.5f); }, lane);
        }
      softmax_row32(scb[h][lane]);
#pragma unroll
      for (int mt = 0; mt < 2; ++mt) {
        v16h a = frag_a_lds(&scb[h][mt * 16][0], 32, lane);
        v16h b = frag_b_lds(&vT[h][0][0], 32, lane);
        v8f c = {};
        c = wmma16(a, b, c);
        frag_store(c, [&](int m, int nn, float v) {
          if (nn < 4) oh[mt * 16 + m][h * 4 + nn] = (_Float16)v; }, lane);
      }
    }
    __syncthreads();

    // ---- Wo projection: M=32, N=20(pad32), K=20(pad32); 4 tiles -> waves 0..3 ----
    {
      const float* wo  = Wo + wb * EMB * EMB;
      const float* bov = bo + wb * EMB;
      if (tid < 32) {
        stage_row<EMB>(wo + tid * EMB, &wh[tid][0], tid < EMB);
        bst[tid] = (tid < EMB) ? bov[tid] : 0.0f;
      }
      __syncthreads();
      if (wave < 4) {
        const int mt = wave & 1, nt = wave >> 1;
        v16h a = frag_a_lds(&oh[mt * 16][0], 32, lane);
        v16h b = frag_b_lds(&wh[nt * 16][0], 32, lane);
        v8f c = {};
        c = wmma16(a, b, c);
        frag_store(c, [&](int m, int nn, float v) {
          pbuf[mt * 16 + m][nt * 16 + nn] = v; }, lane);
      }
      __syncthreads();
    }

    // ---- LN1: c = LN(c + o@Wo^T + bo) ----
    {
      const float* gg = g1 + wb * EMB;
      const float* bb = be1 + wb * EMB;
      if (tid < BB) {
        float r[EMB];
        float mean = 0.0f;
#pragma unroll
        for (int e = 0; e < EMB; ++e) { r[e] = xbf[tid][e] + pbuf[tid][e] + bst[e]; mean += r[e]; }
        mean *= (1.0f / EMB);
        float var = 0.0f;
#pragma unroll
        for (int e = 0; e < EMB; ++e) { const float d = r[e] - mean; var += d * d; }
        var *= (1.0f / EMB);
        const float inv = rsqrtf(var + 1e-5f);
#pragma unroll
        for (int e = 0; e < EMB; ++e) {
          const float o = (r[e] - mean) * inv * gg[e] + bb[e];
          xbf[tid][e] = o;
          xh[tid][e] = (_Float16)o;
        }
      }
      __syncthreads();
    }

    // ---- FFN1: hid = relu(c @ W1^T + b1) : M=32, N=10(pad16), K=20(pad32); waves 0..1 ----
    {
      const float* w1v = W1 + wb * FFN * EMB;
      const float* b1v = b1 + wb * FFN;
      if (tid < 16) {
        stage_row<EMB>(w1v + tid * EMB, &wh[tid][0], tid < FFN);
        bst[tid] = (tid < FFN) ? b1v[tid] : 0.0f;
      }
      __syncthreads();
      if (wave < 2) {
        const int mt = wave;
        v16h a = frag_a_lds(&xh[mt * 16][0], 32, lane);
        v16h b = frag_b_lds(&wh[0][0], 32, lane);
        v8f c = {};
        c = wmma16(a, b, c);
        frag_store(c, [&](int m, int nn, float v) {  // cols 10..15: v=0,b=0 -> stores 0
          hh[mt * 16 + m][nn] = (_Float16)fmaxf(v + bst[nn], 0.0f); }, lane);
      }
      __syncthreads();
    }

    // ---- FFN2: h2 = hid @ W2^T + b2 : M=32, N=20(pad32), K=10(pad32); waves 0..3 ----
    {
      const float* w2v = W2 + wb * EMB * FFN;
      const float* b2v = b2 + wb * EMB;
      if (tid < 32) {
        stage_row<FFN>(w2v + tid * FFN, &wh[tid][0], tid < EMB);
        bst[tid] = (tid < EMB) ? b2v[tid] : 0.0f;
      }
      __syncthreads();
      if (wave < 4) {
        const int mt = wave & 1, nt = wave >> 1;
        v16h a = frag_a_lds(&hh[mt * 16][0], 32, lane);
        v16h b = frag_b_lds(&wh[nt * 16][0], 32, lane);
        v8f c = {};
        c = wmma16(a, b, c);
        frag_store(c, [&](int m, int nn, float v) {
          const int o = nt * 16 + nn;
          pbuf[mt * 16 + m][o] = v + bst[o]; }, lane);
      }
      __syncthreads();
    }

    // ---- LN2: c = LN(c + h2) ----
    {
      const float* gg = g2 + wb * EMB;
      const float* bb = be2 + wb * EMB;
      if (tid < BB) {
        float r[EMB];
        float mean = 0.0f;
#pragma unroll
        for (int e = 0; e < EMB; ++e) { r[e] = xbf[tid][e] + pbuf[tid][e]; mean += r[e]; }
        mean *= (1.0f / EMB);
        float var = 0.0f;
#pragma unroll
        for (int e = 0; e < EMB; ++e) { const float d = r[e] - mean; var += d * d; }
        var *= (1.0f / EMB);
        const float inv = rsqrtf(var + 1e-5f);
#pragma unroll
        for (int e = 0; e < EMB; ++e) {
          const float o = (r[e] - mean) * inv * gg[e] + bb[e];
          xbf[tid][e] = o;
          xh[tid][e] = (_Float16)o;
        }
      }
      __syncthreads();
    }
  }

  for (int e = ecol; e < EMB; e += 8) {
    Y[((size_t)i * BB + srow) * (SEQL * EMB) + n * EMB + e] = xbf[srow][e];
  }
}

// ---------------- Kernel B: per-index Linear(500 -> 10), WMMA over K=512 ----------------

__global__ __launch_bounds__(TB) void perindex_linear_kernel(
    const float* __restrict__ Y, const float* __restrict__ LW,
    const float* __restrict__ Lb, float* __restrict__ sec)
{
  const int i    = blockIdx.x;
  const int tid  = threadIdx.x;
  const int wave = tid >> 5;
  const int lane = tid & 31;

  __shared__ __align__(16) _Float16 yh[BB][512];   // cols < 500
  __shared__ __align__(16) _Float16 lwh[16][512];  // rows < 10, cols < 500

  const float* lw = LW + (size_t)i * SVS * (SEQL * EMB);
  for (int t = tid; t < BB * 512; t += TB) {
    const int s = t >> 9, f = t & 511;
    yh[s][f] = (f < SEQL * EMB) ? (_Float16)Y[((size_t)i * BB + s) * (SEQL * EMB) + f]
                                : (_Float16)0.0f;
  }
  for (int t = tid; t < 16 * 512; t += TB) {
    const int o = t >> 9, f = t & 511;
    lwh[o][f] = (o < SVS && f < SEQL * EMB) ? (_Float16)lw[o * (SEQL * EMB) + f]
                                            : (_Float16)0.0f;
  }
  __syncthreads();

  if (wave < 2) {
    const int mt = wave;
    v8f c = {};
#pragma unroll
    for (int kt = 0; kt < 16; ++kt) {
      v16h a = frag_a_lds(&yh[mt * 16][kt * 32], 512, lane);
      v16h b = frag_b_lds(&lwh[0][kt * 32], 512, lane);
      c = wmma16(a, b, c);
    }
    frag_store(c, [&](int m, int nn, float v) {
      if (nn < SVS)
        sec[((size_t)(mt * 16 + m) * NSEQ + i) * SVS + nn] = v + Lb[i * SVS + nn]; }, lane);
  }
}

// ---------------- Kernel C: second 4-layer encoder slab [S=32 x E=10] ----------------

__global__ __launch_bounds__(TA) void sec_encoder_kernel(
    const float* __restrict__ sec,
    const float* __restrict__ Wqkv, const float* __restrict__ bqkv,
    const float* __restrict__ Wo,   const float* __restrict__ bo,
    const float* __restrict__ W1,   const float* __restrict__ b1,
    const float* __restrict__ W2,   const float* __restrict__ b2,
    const float* __restrict__ g1,   const float* __restrict__ be1,
    const float* __restrict__ g2,   const float* __restrict__ be2,
    float* __restrict__ F)
{
  const int i    = blockIdx.x;
  const int tid  = threadIdx.x;
  const int wave = tid >> 5;
  const int lane = tid & 31;
  const int srow = tid & 31;
  const int ecol = tid >> 5;

  __shared__ __align__(16) _Float16 xh[BB][32];            // cols < 10
  __shared__ float                  xbf[BB][SVS];
  __shared__ __align__(16) _Float16 wh[32][32];
  __shared__ float                  bst[32];
  __shared__ __align__(16) _Float16 qkv32[BB][32];         // raw qkv GEMM result
  __shared__ __align__(16) _Float16 qpad[NHEAD][BB][32];   // cols < 2
  __shared__ __align__(16) _Float16 kpad[NHEAD][BB][32];   // cols < 2
  __shared__ __align__(16) _Float16 vT[NHEAD][16][32];     // rows < 2
  __shared__ __align__(16) _Float16 scb[NHEAD][BB][32];
  __shared__ __align__(16) _Float16 oh[BB][32];            // cols < 10
  __shared__ __align__(16) _Float16 hh[BB][32];            // cols < 16 written
  __shared__ float                  pbuf[BB][16];

  for (int t = tid; t < BB * 32; t += TA) { ((_Float16*)xh)[t] = (_Float16)0.0f;
                                            ((_Float16*)oh)[t] = (_Float16)0.0f;
                                            ((_Float16*)hh)[t] = (_Float16)0.0f; }
  for (int t = tid; t < NHEAD * BB * 32; t += TA) { ((_Float16*)qpad)[t] = (_Float16)0.0f;
                                                    ((_Float16*)kpad)[t] = (_Float16)0.0f; }
  for (int t = tid; t < NHEAD * 16 * 32; t += TA) ((_Float16*)vT)[t] = (_Float16)0.0f;

  for (int e = ecol; e < SVS; e += 8) {
    const float v = sec[((size_t)srow * NSEQ + i) * SVS + e];
    xbf[srow][e] = v;
    xh[srow][e] = (_Float16)v;
  }
  __syncthreads();

#pragma unroll 1
  for (int l = 0; l < NLAY; ++l) {
    // ---- qkv : M=32, N=30(pad32), K=10(pad32); 4 tiles -> waves 0..3 ----
    {
      const float* wq = Wqkv + (size_t)l * 30 * SVS;
      const float* bq = bqkv + (size_t)l * 30;
      if (tid < 32) {
        stage_row<SVS>(wq + tid * SVS, &wh[tid][0], tid < 30);
        bst[tid] = (tid < 30) ? bq[tid] : 0.0f;
      }
      __syncthreads();
      if (wave < 4) {
        const int mt = wave & 1, nt = wave >> 1;
        v16h a = frag_a_lds(&xh[mt * 16][0], 32, lane);
        v16h b = frag_b_lds(&wh[nt * 16][0], 32, lane);
        v8f c = {};
        c = wmma16(a, b, c);
        frag_store(c, [&](int m, int nn, float v) {
          const int o = nt * 16 + nn;
          qkv32[mt * 16 + m][o] = (_Float16)(v + bst[o]); }, lane);
      }
      __syncthreads();
      // repack: q/k head slices are contiguous 4B groups; v scalar transpose (320 elems)
      if (tid < NHEAD * BB) {
        const int h = tid >> 5, row = tid & 31;
        *(unsigned int*)&qpad[h][row][0] = *(const unsigned int*)&qkv32[row][h * 2];
        *(unsigned int*)&kpad[h][row][0] = *(const unsigned int*)&qkv32[row][SVS + h * 2];
      }
      {
        auto vcopy = [&](int t) {
          const int h = t >> 6, rem = t & 63, row = rem >> 1, d = rem & 1;
          vT[h][d][row] = qkv32[row][2 * SVS + h * 2 + d];
        };
        vcopy(tid);
        if (tid < 64) vcopy(tid + 256);
      }
      __syncthreads();
    }

    // ---- attention, D=2, scale 1/sqrt(2): one head per wave ----
    if (wave < NHEAD) {
      const int h = wave;
      const float rs = 0.70710678118654752f;
#pragma unroll
      for (int mt = 0; mt < 2; ++mt)
#pragma unroll
        for (int nt = 0; nt < 2; ++nt) {
          v16h a = frag_a_lds(&qpad[h][mt * 16][0], 32, lane);
          v16h b = frag_b_lds(&kpad[h][nt * 16][0], 32, lane);
          v8f c = {};
          c = wmma16(a, b, c);
          frag_store(c, [&](int m, int nn, float v) {
            scb[h][mt * 16 + m][nt * 16 + nn] = (_Float16)(v * rs); }, lane);
        }
      softmax_row32(scb[h][lane]);
#pragma unroll
      for (int mt = 0; mt < 2; ++mt) {
        v16h a = frag_a_lds(&scb[h][mt * 16][0], 32, lane);
        v16h b = frag_b_lds(&vT[h][0][0], 32, lane);
        v8f c = {};
        c = wmma16(a, b, c);
        frag_store(c, [&](int m, int nn, float v) {
          if (nn < 2) oh[mt * 16 + m][h * 2 + nn] = (_Float16)v; }, lane);
      }
    }
    __syncthreads();

    // ---- Wo : M=32, N=10(pad16), K=10(pad32); waves 0..1 ----
    {
      const float* wo  = Wo + (size_t)l * SVS * SVS;
      const float* bov = bo + (size_t)l * SVS;
      if (tid < 16) {
        stage_row<SVS>(wo + tid * SVS, &wh[tid][0], tid < SVS);
        bst[tid] = (tid < SVS) ? bov[tid] : 0.0f;
      }
      __syncthreads();
      if (wave < 2) {
        const int mt = wave;
        v16h a = frag_a_lds(&oh[mt * 16][0], 32, lane);
        v16h b = frag_b_lds(&wh[0][0], 32, lane);
        v8f c = {};
        c = wmma16(a, b, c);
        frag_store(c, [&](int m, int nn, float v) {
          pbuf[mt * 16 + m][nn] = v; }, lane);
      }
      __syncthreads();
    }

    // ---- LN1 ----
    {
      const float* gg = g1 + (size_t)l * SVS;
      const float* bb = be1 + (size_t)l * SVS;
      if (tid < BB) {
        float r[SVS];
        float mean = 0.0f;
#pragma unroll
        for (int e = 0; e < SVS; ++e) { r[e] = xbf[tid][e] + pbuf[tid][e] + bst[e]; mean += r[e]; }
        mean *= (1.0f / SVS);
        float var = 0.0f;
#pragma unroll
        for (int e = 0; e < SVS; ++e) { const float d = r[e] - mean; var += d * d; }
        var *= (1.0f / SVS);
        const float inv = rsqrtf(var + 1e-5f);
#pragma unroll
        for (int e = 0; e < SVS; ++e) {
          const float o = (r[e] - mean) * inv * gg[e] + bb[e];
          xbf[tid][e] = o;
          xh[tid][e] = (_Float16)o;
        }
      }
      __syncthreads();
    }

    // ---- FFN1 : M=32, N=10(pad16), K=10(pad32); waves 0..1 ----
    {
      const float* w1v = W1 + (size_t)l * FFN * SVS;
      const float* b1v = b1 + (size_t)l * FFN;
      if (tid < 16) {
        stage_row<SVS>(w1v + tid * SVS, &wh[tid][0], tid < FFN);
        bst[tid] = (tid < FFN) ? b1v[tid] : 0.0f;
      }
      __syncthreads();
      if (wave < 2) {
        const int mt = wave;
        v16h a = frag_a_lds(&xh[mt * 16][0], 32, lane);
        v16h b = frag_b_lds(&wh[0][0], 32, lane);
        v8f c = {};
        c = wmma16(a, b, c);
        frag_store(c, [&](int m, int nn, float v) {
          hh[mt * 16 + m][nn] = (_Float16)fmaxf(v + bst[nn], 0.0f); }, lane);
      }
      __syncthreads();
    }

    // ---- FFN2 : M=32, N=10(pad16), K=10(pad32); waves 0..1 ----
    {
      const float* w2v = W2 + (size_t)l * SVS * FFN;
      const float* b2v = b2 + (size_t)l * SVS;
      if (tid < 16) {
        stage_row<FFN>(w2v + tid * FFN, &wh[tid][0], tid < SVS);
        bst[tid] = (tid < SVS) ? b2v[tid] : 0.0f;
      }
      __syncthreads();
      if (wave < 2) {
        const int mt = wave;
        v16h a = frag_a_lds(&hh[mt * 16][0], 32, lane);
        v16h b = frag_b_lds(&wh[0][0], 32, lane);
        v8f c = {};
        c = wmma16(a, b, c);
        frag_store(c, [&](int m, int nn, float v) {
          pbuf[mt * 16 + m][nn] = v + bst[nn]; }, lane);
      }
      __syncthreads();
    }

    // ---- LN2 ----
    {
      const float* gg = g2 + (size_t)l * SVS;
      const float* bb = be2 + (size_t)l * SVS;
      if (tid < BB) {
        float r[SVS];
        float mean = 0.0f;
#pragma unroll
        for (int e = 0; e < SVS; ++e) { r[e] = xbf[tid][e] + pbuf[tid][e]; mean += r[e]; }
        mean *= (1.0f / SVS);
        float var = 0.0f;
#pragma unroll
        for (int e = 0; e < SVS; ++e) { const float d = r[e] - mean; var += d * d; }
        var *= (1.0f / SVS);
        const float inv = rsqrtf(var + 1e-5f);
#pragma unroll
        for (int e = 0; e < SVS; ++e) {
          const float o = (r[e] - mean) * inv * gg[e] + bb[e];
          xbf[tid][e] = o;
          xh[tid][e] = (_Float16)o;
        }
      }
      __syncthreads();
    }
  }

  for (int e = ecol; e < SVS; e += 8) {
    F[(size_t)srow * (NSEQ * SVS) + i * SVS + e] = xbf[srow][e];
  }
}

// ---------------- Kernel D: head (4300 -> 8 -> 2) ----------------

__global__ __launch_bounds__(TB) void head_kernel(
    const float* __restrict__ F, const float* __restrict__ plW, const float* __restrict__ plb,
    const float* __restrict__ llW, const float* __restrict__ llb, float* __restrict__ out)
{
  __shared__ float t8[BB][8];
  const int tid = threadIdx.x;
  for (int task = tid; task < BB * 8; task += TB) {
    const int b2 = task >> 3, o = task & 7;
    const float4* fr = (const float4*)(F + (size_t)b2 * (NSEQ * SVS));
    const float4* wr = (const float4*)(plW + (size_t)o * (NSEQ * SVS));
    float s = plb[o];
    for (int f = 0; f < (NSEQ * SVS) / 4; ++f) {
      const float4 a = fr[f], b = wr[f];
      s += a.x * b.x + a.y * b.y + a.z * b.z + a.w * b.w;
    }
    t8[b2][o] = s;
  }
  __syncthreads();
  for (int task = tid; task < BB * 2; task += TB) {
    const int b2 = task >> 1, c2 = task & 1;
    float s = llb[c2];
#pragma unroll
    for (int o = 0; o < 8; ++o) s += t8[b2][o] * llW[c2 * 8 + o];
    out[b2 * 2 + c2] = s;
  }
}

// ---------------- launch ----------------

extern "C" void kernel_launch(void* const* d_in, const int* in_sizes, int n_in,
                              void* d_out, int out_size, void* d_ws, size_t ws_size,
                              hipStream_t stream) {
  (void)in_sizes; (void)n_in; (void)out_size; (void)ws_size;
  const float* X        = (const float*)d_in[0];
  const float* pt_Wqkv  = (const float*)d_in[1];
  const float* pt_bqkv  = (const float*)d_in[2];
  const float* pt_Wo    = (const float*)d_in[3];
  const float* pt_bo    = (const float*)d_in[4];
  const float* pt_W1    = (const float*)d_in[5];
  const float* pt_b1    = (const float*)d_in[6];
  const float* pt_W2    = (const float*)d_in[7];
  const float* pt_b2    = (const float*)d_in[8];
  const float* pt_g1    = (const float*)d_in[9];
  const float* pt_beta1 = (const float*)d_in[10];
  const float* pt_g2    = (const float*)d_in[11];
  const float* pt_beta2 = (const float*)d_in[12];
  const float* lin_W    = (const float*)d_in[13];
  const float* lin_b    = (const float*)d_in[14];
  const float* st_Wqkv  = (const float*)d_in[15];
  const float* st_bqkv  = (const float*)d_in[16];
  const float* st_Wo    = (const float*)d_in[17];
  const float* st_bo    = (const float*)d_in[18];
  const float* st_W1    = (const float*)d_in[19];
  const float* st_b1    = (const float*)d_in[20];
  const float* st_W2    = (const float*)d_in[21];
  const float* st_b2    = (const float*)d_in[22];
  const float* st_g1    = (const float*)d_in[23];
  const float* st_beta1 = (const float*)d_in[24];
  const float* st_g2    = (const float*)d_in[25];
  const float* st_beta2 = (const float*)d_in[26];
  const float* pl_W     = (const float*)d_in[27];
  const float* pl_b     = (const float*)d_in[28];
  const float* ll_W     = (const float*)d_in[29];
  const float* ll_b     = (const float*)d_in[30];

  float* ws  = (float*)d_ws;
  float* Y   = ws;                                   // 430*32*500 f32
  float* sec = Y + (size_t)NSEQ * BB * (SEQL * EMB); // 32*430*10  f32
  float* F   = sec + (size_t)BB * NSEQ * SVS;        // 32*4300    f32

  dim3 gA(NSEQ, SEQL);
  win_encoder_kernel<<<gA, TA, 0, stream>>>(
      X, pt_Wqkv, pt_bqkv, pt_Wo, pt_bo, pt_W1, pt_b1, pt_W2, pt_b2,
      pt_g1, pt_beta1, pt_g2, pt_beta2, Y);

  perindex_linear_kernel<<<NSEQ, TB, 0, stream>>>(Y, lin_W, lin_b, sec);

  sec_encoder_kernel<<<NSEQ, TA, 0, stream>>>(
      sec, st_Wqkv, st_bqkv, st_Wo, st_bo, st_W1, st_b1, st_W2, st_b2,
      st_g1, st_beta1, st_g2, st_beta2, F);

  head_kernel<<<1, TB, 0, stream>>>(F, pl_W, pl_b, ll_W, ll_b, (float*)d_out);
}
